// Net_66726611911314
// MI455X (gfx1250) — matrix operand order repaired and checked by reading, not compiled, not run on hardware
//
#include <hip/hip_runtime.h>
#include <math.h>

#define B_    512
#define BAG_  32
#define H_    1024
#define C_    53
#define EPS_  1e-8f

typedef __attribute__((ext_vector_type(16))) _Float16 v16h;
typedef __attribute__((ext_vector_type(8)))  float    v8f;

// ---------------------------------------------------------------------------
// Kernel 1: per-bag attention + norms.
// block b: scores[j]=rel.sent[b,j]/32 -> softmax -> bag_reps[b,:]
// also nr[b,j]=|sent|, na[b,j]=|aug|, pd[b,j]=sent.aug, nb[b]=|bag|
// ---------------------------------------------------------------------------
__global__ void __launch_bounds__(256) k_attn(
    const float* __restrict__ sent, const float* __restrict__ aug,
    const float* __restrict__ rel_table, const int* __restrict__ labels,
    float* __restrict__ bag_reps, float* __restrict__ nr,
    float* __restrict__ na, float* __restrict__ pd, float* __restrict__ nb) {
  const int b = blockIdx.x;
  const int tid = threadIdx.x, lane = tid & 31, wav = tid >> 5;
  __shared__ float scores[BAG_];
  __shared__ float alphas[BAG_];
  __shared__ float red8[8];
  const float* relv = rel_table + (size_t)labels[b] * H_;
  const float* sb = sent + (size_t)b * BAG_ * H_;
  const float* ab = aug  + (size_t)b * BAG_ * H_;

  for (int j = wav; j < BAG_; j += 8) {
    const float* sj = sb + j * H_;
    const float* aj = ab + j * H_;
    float s_rel = 0.f, s_rr = 0.f, s_aa = 0.f, s_ra = 0.f;
    for (int h = lane; h < H_; h += 32) {
      float sv = sj[h], av = aj[h], rv = relv[h];
      s_rel += rv * sv; s_rr += sv * sv; s_aa += av * av; s_ra += sv * av;
    }
    for (int off = 16; off > 0; off >>= 1) {
      s_rel += __shfl_xor(s_rel, off);
      s_rr  += __shfl_xor(s_rr, off);
      s_aa  += __shfl_xor(s_aa, off);
      s_ra  += __shfl_xor(s_ra, off);
    }
    if (lane == 0) {
      scores[j] = s_rel * (1.0f / 32.0f);   // / sqrt(H)
      nr[b * BAG_ + j] = sqrtf(s_rr);
      na[b * BAG_ + j] = sqrtf(s_aa);
      pd[b * BAG_ + j] = s_ra;
    }
  }
  __syncthreads();
  if (tid < 32) {                           // softmax over 32 scores (wave 0)
    float v = scores[tid];
    float m = v;
    for (int off = 16; off > 0; off >>= 1) m = fmaxf(m, __shfl_xor(m, off));
    float e = __expf(v - m);
    float s = e;
    for (int off = 16; off > 0; off >>= 1) s += __shfl_xor(s, off);
    alphas[tid] = e / s;
  }
  __syncthreads();
  float ssq = 0.f;
  for (int h = tid; h < H_; h += 256) {
    float acc = 0.f;
    for (int j = 0; j < BAG_; ++j) acc += alphas[j] * sb[j * H_ + h];
    bag_reps[(size_t)b * H_ + h] = acc;
    ssq += acc * acc;
  }
  for (int off = 16; off > 0; off >>= 1) ssq += __shfl_xor(ssq, off);
  if (lane == 0) red8[wav] = ssq;
  __syncthreads();
  if (tid == 0) {
    float t = 0.f;
    for (int w = 0; w < 8; ++w) t += red8[w];   // fixed order: deterministic
    nb[b] = sqrtf(t);
  }
}

// ---------------------------------------------------------------------------
// Kernel 2: dots = sent(16384x1024) @ bag_reps^T(1024x512) via WMMA f16,
// fused epilogue exp(dots/(nb*nr*T)) -> deterministic per-rowblock partials.
// grid = (8 col strips of 64, 64 row blocks of 256). block = 8 waves.
// wave w owns one full bag (rows r0+w*32 .. +31), tiles 2M x 4N.
// ---------------------------------------------------------------------------
#define KC_   128           // K chunk staged in LDS
#define KCP_  136           // padded halves per column (16B-aligned stride)
#define CBLK_ 64
#define RBLK_ 256

__device__ __forceinline__ v16h load_a_frag(const float* __restrict__ p) {
  // lane layout (16-bit A 16x32): halves 0..7 = K kbase..kbase+7,
  // halves 8..15 = K kbase+16..kbase+23 (p already offset by row+kbase)
  union { v16h v; _Float16 h[16]; } u;
  float4 f0 = *(const float4*)(p);
  float4 f1 = *(const float4*)(p + 4);
  float4 f2 = *(const float4*)(p + 16);
  float4 f3 = *(const float4*)(p + 20);
  u.h[0]=(_Float16)f0.x;  u.h[1]=(_Float16)f0.y;  u.h[2]=(_Float16)f0.z;  u.h[3]=(_Float16)f0.w;
  u.h[4]=(_Float16)f1.x;  u.h[5]=(_Float16)f1.y;  u.h[6]=(_Float16)f1.z;  u.h[7]=(_Float16)f1.w;
  u.h[8]=(_Float16)f2.x;  u.h[9]=(_Float16)f2.y;  u.h[10]=(_Float16)f2.z; u.h[11]=(_Float16)f2.w;
  u.h[12]=(_Float16)f3.x; u.h[13]=(_Float16)f3.y; u.h[14]=(_Float16)f3.z; u.h[15]=(_Float16)f3.w;
  return u.v;
}

__global__ void __launch_bounds__(256) k_gemm_neg(
    const float* __restrict__ sent, const float* __restrict__ bag_reps,
    const float* __restrict__ nr, const float* __restrict__ nb,
    const float* __restrict__ temp,
    float* __restrict__ negpart, float* __restrict__ diag) {
  __shared__ _Float16 ldsB[CBLK_ * KCP_];   // 17,408 B
  __shared__ float    ldsW[4 * 2048];       // 32,768 B: 4 e-value slices
  const int tid = threadIdx.x, lane = tid & 31, wav = tid >> 5;
  const int lo = lane & 15, hi = lane >> 4;
  const int c0 = blockIdx.x * CBLK_;
  const int r0 = blockIdx.y * RBLK_;

  v8f acc[2][4];
  const v8f vz = {0.f, 0.f, 0.f, 0.f, 0.f, 0.f, 0.f, 0.f};
  for (int mt = 0; mt < 2; ++mt)
    for (int nt = 0; nt < 4; ++nt) acc[mt][nt] = vz;

  const float* pa0 = sent + (size_t)(r0 + wav * 32 + lo) * H_;
  const float* pa1 = pa0 + 16 * H_;

  for (int kc = 0; kc < H_; kc += KC_) {
    __syncthreads();
    {   // stage B chunk: 64 cols x 128 K, f32 -> f16 LDS
      int col = tid >> 2;
      int kq = (tid & 3) * 32;
      const float* src = bag_reps + (size_t)(c0 + col) * H_ + kc + kq;
      _Float16* dst = ldsB + col * KCP_ + kq;
      for (int i = 0; i < 32; i += 4) {
        float4 f = *(const float4*)(src + i);
        dst[i + 0] = (_Float16)f.x; dst[i + 1] = (_Float16)f.y;
        dst[i + 2] = (_Float16)f.z; dst[i + 3] = (_Float16)f.w;
      }
    }
    __syncthreads();
    for (int ks = 0; ks < KC_; ks += 32) {
      union { v16h v; float4 f[2]; } bu[4];
      for (int nt = 0; nt < 4; ++nt) {
        // B layout: lane=col lo (both halves), halves 0..7 = K 8*hi..+7, halves 8..15 = K 16+8*hi..+7
        int colLocal = nt * 16 + lo;
        int kLocal = ks + hi * 8;
        const float4* lp =
            (const float4*)((const char*)ldsB + ((size_t)colLocal * KCP_ + kLocal) * 2);
        bu[nt].f[0] = lp[0];
        bu[nt].f[1] = lp[2];
      }
      int gk = kc + ks + hi * 8;
      v16h a0 = load_a_frag(pa0 + gk);
      v16h a1 = load_a_frag(pa1 + gk);
      for (int nt = 0; nt < 4; ++nt) {
        acc[0][nt] = __builtin_amdgcn_wmma_f32_16x16x32_f16(
            false, a0, false, bu[nt].v, (short)0, acc[0][nt], false, false);
        acc[1][nt] = __builtin_amdgcn_wmma_f32_16x16x32_f16(
            false, a1, false, bu[nt].v, (short)0, acc[1][nt], false, false);
      }
    }
  }

  // ---- epilogue: e = exp(dots/(nb[c]*nr[b',j]*T)); deterministic reduce ----
  const float invT = 1.0f / temp[0];
  const int bagIdx = blockIdx.y * 8 + wav;            // this wave's b'
  const int slice = wav & 3;
  auto epi = [&](bool addMode) {
    for (int mt = 0; mt < 2; ++mt) {
      for (int nt = 0; nt < 4; ++nt) {
        int c = c0 + nt * 16 + lo;
        float nbc = nb[c];
        for (int r = 0; r < 8; ++r) {
          int j = mt * 16 + hi * 8 + r;                // C/D layout row
          float val = acc[mt][nt][r];
          float den = fmaxf(nbc * nr[bagIdx * BAG_ + j], EPS_);
          float e = __expf(val / den * invT);
          if (c == bagIdx) diag[bagIdx * BAG_ + j] = e;  // axis_sim term
          int idx = slice * 2048 + (nt * 16 + lo) * BAG_ + j;
          if (addMode) ldsW[idx] += e; else ldsW[idx] = e;
        }
      }
    }
  };
  __syncthreads();
  if (wav < 4) epi(false);        // waves 0-3 write their slice
  __syncthreads();
  if (wav >= 4) epi(true);        // waves 4-7 add onto slice (wav-4)
  __syncthreads();
  for (int idx = tid; idx < 2048; idx += 256) {        // fixed-order sum
    float s = ldsW[idx] + ldsW[2048 + idx] + ldsW[4096 + idx] + ldsW[6144 + idx];
    int cl = idx >> 5, j = idx & 31;
    negpart[(size_t)blockIdx.y * (B_ * BAG_) + (size_t)(c0 + cl) * BAG_ + j] = s;
  }
}

// ---------------------------------------------------------------------------
// Kernel 3: bag_out = bag_reps @ cls_w^T + cls_b
// ---------------------------------------------------------------------------
__global__ void __launch_bounds__(256) k_bagout(
    const float* __restrict__ bag_reps, const float* __restrict__ cls_w,
    const float* __restrict__ cls_b, float* __restrict__ out) {
  const int b = blockIdx.x;
  const int tid = threadIdx.x, lane = tid & 31, wav = tid >> 5;
  __shared__ float bl[H_];
  for (int h = tid; h < H_; h += 256) bl[h] = bag_reps[(size_t)b * H_ + h];
  __syncthreads();
  for (int c = wav; c < C_; c += 8) {
    const float* wrow = cls_w + (size_t)c * H_;
    float s = 0.f;
    for (int h = lane; h < H_; h += 32) s += bl[h] * wrow[h];
    for (int off = 16; off > 0; off >>= 1) s += __shfl_xor(s, off);
    if (lane == 0) out[b * C_ + c] = s + cls_b[c];
  }
}

// ---------------------------------------------------------------------------
// Kernel 4: per-element loss + per-block partial sums (fixed-order reductions)
// ---------------------------------------------------------------------------
__global__ void __launch_bounds__(256) k_loss_part(
    const float* __restrict__ negpart, const float* __restrict__ diag,
    const float* __restrict__ pd, const float* __restrict__ nr,
    const float* __restrict__ na, const float* __restrict__ temp,
    float* __restrict__ blockpart) {
  __shared__ float sdata[256];
  const int tid = threadIdx.x;
  const int i = blockIdx.x * 256 + tid;               // 0..16383
  float s = 0.f;
  for (int rb = 0; rb < 64; ++rb) s += negpart[(size_t)rb * (B_ * BAG_) + i];
  float neg = s - diag[i];
  float invT = 1.0f / temp[0];
  float pos_sim = pd[i] / fmaxf(nr[i] * na[i], EPS_);
  float pos = __expf(pos_sim * invT);
  sdata[tid] = -__logf(pos / (pos + neg));
  __syncthreads();
  for (int st = 128; st > 0; st >>= 1) {
    if (tid < st) sdata[tid] += sdata[tid + st];
    __syncthreads();
  }
  if (tid == 0) blockpart[blockIdx.x] = sdata[0];
}

__global__ void k_loss_final(const float* __restrict__ blockpart,
                             float* __restrict__ out_scalar) {
  if (threadIdx.x == 0) {
    float t = 0.f;
    for (int i = 0; i < 64; ++i) t += blockpart[i];
    out_scalar[0] = t * (1.0f / (float)(B_ * BAG_));
  }
}

// ---------------------------------------------------------------------------
extern "C" void kernel_launch(void* const* d_in, const int* in_sizes, int n_in,
                              void* d_out, int out_size, void* d_ws, size_t ws_size,
                              hipStream_t stream) {
  const float* sent      = (const float*)d_in[0];
  const float* aug       = (const float*)d_in[1];
  const float* rel_table = (const float*)d_in[2];
  const float* cls_w     = (const float*)d_in[3];
  const float* cls_b     = (const float*)d_in[4];
  const int*   labels    = (const int*)d_in[5];
  const float* temp      = (const float*)d_in[6];
  float* out = (float*)d_out;

  float* ws        = (float*)d_ws;
  float* bag_reps  = ws;                      // 512*1024
  float* nr        = bag_reps + (size_t)B_ * H_;     // 16384
  float* na        = nr + B_ * BAG_;
  float* pd        = na + B_ * BAG_;
  float* nb        = pd + B_ * BAG_;          // 512
  float* diag      = nb + B_;                 // 16384
  float* negpart   = diag + B_ * BAG_;        // 64*16384
  float* blockpart = negpart + (size_t)64 * B_ * BAG_;  // 64

  k_attn<<<B_, 256, 0, stream>>>(sent, aug, rel_table, labels,
                                 bag_reps, nr, na, pd, nb);
  dim3 g2(B_ / CBLK_, (B_ * BAG_) / RBLK_);   // (8, 64)
  k_gemm_neg<<<g2, 256, 0, stream>>>(sent, bag_reps, nr, nb, temp,
                                     negpart, diag);
  k_bagout<<<B_, 256, 0, stream>>>(bag_reps, cls_w, cls_b, out);
  k_loss_part<<<64, 256, 0, stream>>>(negpart, diag, pd, nr, na, temp,
                                      blockpart);
  k_loss_final<<<1, 32, 0, stream>>>(blockpart, out + B_ * C_);
}
